// Self_Attn_78365973282833
// MI455X (gfx1250) — compile-verified, matrix operand order checked by reading, not run on hardware
//
#include <hip/hip_runtime.h>

typedef __attribute__((ext_vector_type(16))) _Float16 v16h;
typedef __attribute__((ext_vector_type(8)))  _Float16 v8h;
typedef __attribute__((ext_vector_type(8)))  float    v8f;

// ---------------------------------------------------------------------------
// Fragment loaders for v_wmma_f32_16x16x32_f16 (wave32).
// A fragment: row (lane&15), K elements: e<8 -> k0 + h*8 + e
//                                        e>=8 -> k0 + 16 + h*8 + (e-8), h = lane>>4.
// B fragment is symmetric with "row" = column index, so both use the same
// loader as long as the operand is stored K-contiguous ([rows][K] row-major).
// Each fragment = two 16-byte loads per lane.
// ---------------------------------------------------------------------------
static __device__ inline v16h frag_h(const _Float16* __restrict__ base,
                                     int row0, int k0, int ld) {
    int lane = threadIdx.x & 31;
    int m = lane & 15, hh = lane >> 4;
    const _Float16* p = base + (size_t)(row0 + m) * ld + (k0 + hh * 8);
    v8h lo = *(const v8h*)p;
    v8h hi = *(const v8h*)(p + 16);
    return __builtin_shufflevector(lo, hi, 0,1,2,3,4,5,6,7,8,9,10,11,12,13,14,15);
}

// Same fragment but sourced from fp32 memory (attention matrix), converting
// to f16 on the fly.
static __device__ inline v16h frag_f32(const float* __restrict__ base,
                                       int row0, int k0, int ld) {
    int lane = threadIdx.x & 31;
    int m = lane & 15, hh = lane >> 4;
    const float* p = base + (size_t)(row0 + m) * ld + (k0 + hh * 8);
    v16h f;
#pragma unroll
    for (int i = 0; i < 8; ++i) {
        f[i]     = (_Float16)p[i];
        f[i + 8] = (_Float16)p[16 + i];
    }
    return f;
}

#define WMMA_F16(A, Bv, Cacc) \
    __builtin_amdgcn_wmma_f32_16x16x32_f16(false, (A), false, (Bv), (short)0, (Cacc), false, false)

// ---------------------------------------------------------------------------
// Kernel: 32x32-per-block transpose + f32->f16 convert:  xh[b][n][c] = x[b][c][n]
// ---------------------------------------------------------------------------
__global__ __launch_bounds__(256)
void k_transpose_cvt(const float* __restrict__ x, _Float16* __restrict__ xh,
                     int Cdim, int Ndim) {
    __shared__ float tile[32][33];
    int b = blockIdx.z;
    const float* xb = x + (size_t)b * Cdim * Ndim;
    _Float16*    xo = xh + (size_t)b * Cdim * Ndim;
    int n0 = blockIdx.x * 32, c0 = blockIdx.y * 32;
    int tx = threadIdx.x, ty = threadIdx.y;
#pragma unroll
    for (int i = 0; i < 4; ++i) {
        int cc = ty + i * 8;
        tile[cc][tx] = xb[(size_t)(c0 + cc) * Ndim + (n0 + tx)];
    }
    __syncthreads();
#pragma unroll
    for (int i = 0; i < 4; ++i) {
        int nn = ty + i * 8;
        xo[(size_t)(n0 + nn) * Cdim + (c0 + tx)] = (_Float16)tile[tx][nn];
    }
}

__global__ void k_cvt(const float* __restrict__ src, _Float16* __restrict__ dst, int n) {
    int i = blockIdx.x * 256 + threadIdx.x;
    if (i < n) dst[i] = (_Float16)src[i];
}

// ---------------------------------------------------------------------------
// Generic projection GEMM:  D[row][col] (f16) = sum_k A[row][k]*Bt[col][k] + bias
// A: [M x K] f16, Bt: [Ncol x K] f16 (both K-contiguous).  Block = 8 waves,
// block tile 128(M) x 64(N); each wave owns a 32x32 tile (2x2 WMMA accums).
// ---------------------------------------------------------------------------
__global__ __launch_bounds__(256)
void k_gemm_f16(const _Float16* __restrict__ A, const _Float16* __restrict__ Bt,
                const float* __restrict__ bias, _Float16* __restrict__ Dm,
                int K, int ldD,
                long long aBatch, long long bBatch, long long dBatch, int biasPerRow) {
    int b = blockIdx.z;
    A  += (size_t)b * aBatch;
    Bt += (size_t)b * bBatch;
    Dm += (size_t)b * dBatch;
    int w = threadIdx.x >> 5, lane = threadIdx.x & 31;
    int row0 = blockIdx.x * 128 + (w & 3) * 32;
    int col0 = blockIdx.y * 64  + (w >> 2) * 32;
    v8f acc[2][2] = {};
    for (int k = 0; k < K; k += 32) {
        v16h a0 = frag_h(A,  row0,      k, K);
        v16h a1 = frag_h(A,  row0 + 16, k, K);
        v16h b0 = frag_h(Bt, col0,      k, K);
        v16h b1 = frag_h(Bt, col0 + 16, k, K);
        acc[0][0] = WMMA_F16(a0, b0, acc[0][0]);
        acc[0][1] = WMMA_F16(a0, b1, acc[0][1]);
        acc[1][0] = WMMA_F16(a1, b0, acc[1][0]);
        acc[1][1] = WMMA_F16(a1, b1, acc[1][1]);
    }
    int cn = lane & 15, hh = lane >> 4;
#pragma unroll
    for (int i = 0; i < 2; ++i)
#pragma unroll
        for (int j = 0; j < 2; ++j) {
            int col = col0 + j * 16 + cn;
#pragma unroll
            for (int r = 0; r < 8; ++r) {
                int row = row0 + i * 16 + hh * 8 + r;
                float v = acc[i][j][r] + (biasPerRow ? bias[row] : bias[col]);
                Dm[(size_t)row * ldD + col] = (_Float16)v;
            }
        }
}

// ---------------------------------------------------------------------------
// Energy GEMM: E[m][n] (f32) = sum_d Q[m][d] * Kt[n][d].  K = D = 64.
// ---------------------------------------------------------------------------
__global__ __launch_bounds__(256)
void k_energy(const _Float16* __restrict__ Q, const _Float16* __restrict__ Kt,
              float* __restrict__ E, int N, int Dk,
              long long qBatch, long long kBatch, long long eBatch) {
    int b = blockIdx.z;
    Q  += (size_t)b * qBatch;
    Kt += (size_t)b * kBatch;
    E  += (size_t)b * eBatch;
    int w = threadIdx.x >> 5, lane = threadIdx.x & 31;
    int row0 = blockIdx.x * 128 + (w & 3) * 32;
    int col0 = blockIdx.y * 64  + (w >> 2) * 32;
    v8f acc[2][2] = {};
    for (int k = 0; k < Dk; k += 32) {
        v16h a0 = frag_h(Q,  row0,      k, Dk);
        v16h a1 = frag_h(Q,  row0 + 16, k, Dk);
        v16h b0 = frag_h(Kt, col0,      k, Dk);
        v16h b1 = frag_h(Kt, col0 + 16, k, Dk);
        acc[0][0] = WMMA_F16(a0, b0, acc[0][0]);
        acc[0][1] = WMMA_F16(a0, b1, acc[0][1]);
        acc[1][0] = WMMA_F16(a1, b0, acc[1][0]);
        acc[1][1] = WMMA_F16(a1, b1, acc[1][1]);
    }
    int cn = lane & 15, hh = lane >> 4;
#pragma unroll
    for (int i = 0; i < 2; ++i)
#pragma unroll
        for (int j = 0; j < 2; ++j) {
            int col = col0 + j * 16 + cn;
#pragma unroll
            for (int r = 0; r < 8; ++r) {
                int row = row0 + i * 16 + hh * 8 + r;
                E[(size_t)row * N + col] = acc[i][j][r];
            }
        }
}

// ---------------------------------------------------------------------------
// Row softmax over N=4096 columns.  One block (256 threads / 8 waves) per row.
// wave32 shuffle reduction + LDS cross-wave reduction.
// ---------------------------------------------------------------------------
__global__ __launch_bounds__(256)
void k_softmax(float* __restrict__ attn, int N) {
    float* row = attn + (size_t)blockIdx.x * N;
    int t = threadIdx.x;
    float vals[16];
    float mx = -3.0e38f;
#pragma unroll
    for (int i = 0; i < 16; ++i) {
        vals[i] = row[t + 256 * i];
        mx = fmaxf(mx, vals[i]);
    }
#pragma unroll
    for (int o = 16; o > 0; o >>= 1) mx = fmaxf(mx, __shfl_xor(mx, o, 32));
    __shared__ float red[8];
    if ((t & 31) == 0) red[t >> 5] = mx;
    __syncthreads();
    float gmx = red[0];
#pragma unroll
    for (int i = 1; i < 8; ++i) gmx = fmaxf(gmx, red[i]);
    __syncthreads();

    float s = 0.f;
#pragma unroll
    for (int i = 0; i < 16; ++i) {
        vals[i] = __expf(vals[i] - gmx);
        s += vals[i];
    }
#pragma unroll
    for (int o = 16; o > 0; o >>= 1) s += __shfl_xor(s, o, 32);
    if ((t & 31) == 0) red[t >> 5] = s;
    __syncthreads();
    float tot = 0.f;
#pragma unroll
    for (int i = 0; i < 8; ++i) tot += red[i];
    float inv = 1.f / tot;
#pragma unroll
    for (int i = 0; i < 16; ++i) row[t + 256 * i] = vals[i] * inv;
}

// ---------------------------------------------------------------------------
// Output GEMM: out[b][c][m] = gamma * sum_n attn[m][n]*v[c][n] + x[b][c][m]
// A = attn (fp32, converted per-fragment), Bt = vh [C][N] f16.
// ---------------------------------------------------------------------------
__global__ __launch_bounds__(256)
void k_out(const float* __restrict__ attn, const _Float16* __restrict__ Vh,
           const float* __restrict__ x, const float* __restrict__ gamma,
           float* __restrict__ out, int N, int Cdim,
           long long aBatch, long long vBatch, long long xBatch, long long oBatch) {
    int b = blockIdx.z;
    attn += (size_t)b * aBatch;
    Vh   += (size_t)b * vBatch;
    x    += (size_t)b * xBatch;
    out  += (size_t)b * oBatch;
    int w = threadIdx.x >> 5, lane = threadIdx.x & 31;
    int row0 = blockIdx.x * 128 + (w & 3) * 32;   // m
    int col0 = blockIdx.y * 64  + (w >> 2) * 32;  // c
    v8f acc[2][2] = {};
    for (int k = 0; k < N; k += 32) {
        v16h a0 = frag_f32(attn, row0,      k, N);
        v16h a1 = frag_f32(attn, row0 + 16, k, N);
        v16h b0 = frag_h(Vh, col0,      k, N);
        v16h b1 = frag_h(Vh, col0 + 16, k, N);
        acc[0][0] = WMMA_F16(a0, b0, acc[0][0]);
        acc[0][1] = WMMA_F16(a0, b1, acc[0][1]);
        acc[1][0] = WMMA_F16(a1, b0, acc[1][0]);
        acc[1][1] = WMMA_F16(a1, b1, acc[1][1]);
    }
    float g = gamma[0];
    int cn = lane & 15, hh = lane >> 4;
#pragma unroll
    for (int i = 0; i < 2; ++i)
#pragma unroll
        for (int j = 0; j < 2; ++j) {
            int c = col0 + j * 16 + cn;
#pragma unroll
            for (int r = 0; r < 8; ++r) {
                int m = row0 + i * 16 + hh * 8 + r;
                size_t idx = (size_t)c * N + m;
                out[idx] = g * acc[i][j][r] + x[idx];
            }
        }
}

// ---------------------------------------------------------------------------
extern "C" void kernel_launch(void* const* d_in, const int* in_sizes, int n_in,
                              void* d_out, int out_size, void* d_ws, size_t ws_size,
                              hipStream_t stream) {
    (void)in_sizes; (void)n_in; (void)out_size; (void)ws_size;
    const float* x     = (const float*)d_in[0];
    const float* Wq    = (const float*)d_in[1];
    const float* bq    = (const float*)d_in[2];
    const float* Wk    = (const float*)d_in[3];
    const float* bk    = (const float*)d_in[4];
    const float* Wv    = (const float*)d_in[5];
    const float* bv    = (const float*)d_in[6];
    const float* gamma = (const float*)d_in[7];

    constexpr int B = 4, C = 512, Wd = 64, Hd = 64, N = Wd * Hd, D = C / 8;

    float* out  = (float*)d_out;              // [B][C][N]
    float* attn = out + (size_t)B * C * N;    // [B][N][N]

    char* ws = (char*)d_ws;
    _Float16* xh  = (_Float16*)ws; ws += (size_t)B * N * C * 2;  // x^T  f16 [B][N][C]
    _Float16* qh  = (_Float16*)ws; ws += (size_t)B * N * D * 2;  // q    f16 [B][N][D]
    _Float16* kth = (_Float16*)ws; ws += (size_t)B * N * D * 2;  // k^T  f16 [B][N][D]
    _Float16* vh  = (_Float16*)ws; ws += (size_t)B * C * N * 2;  // v    f16 [B][C][N]
    _Float16* Wqh = (_Float16*)ws; ws += (size_t)D * C * 2;
    _Float16* Wkh = (_Float16*)ws; ws += (size_t)D * C * 2;
    _Float16* Wvh = (_Float16*)ws; ws += (size_t)C * C * 2;

    // 0) layout conversion to f16 / K-contiguous
    k_transpose_cvt<<<dim3(N / 32, C / 32, B), dim3(32, 8), 0, stream>>>(x, xh, C, N);
    k_cvt<<<(D * C + 255) / 256, 256, 0, stream>>>(Wq, Wqh, D * C);
    k_cvt<<<(D * C + 255) / 256, 256, 0, stream>>>(Wk, Wkh, D * C);
    k_cvt<<<(C * C + 255) / 256, 256, 0, stream>>>(Wv, Wvh, C * C);

    // 1) projections: q/kT = xh * W^T + b  (bias per col),  v = Wv * x + b (bias per row)
    k_gemm_f16<<<dim3(N / 128, D / 64, B), 256, 0, stream>>>(
        xh, Wqh, bq, qh, C, D, (long long)N * C, 0, (long long)N * D, 0);
    k_gemm_f16<<<dim3(N / 128, D / 64, B), 256, 0, stream>>>(
        xh, Wkh, bk, kth, C, D, (long long)N * C, 0, (long long)N * D, 0);
    k_gemm_f16<<<dim3(C / 128, N / 64, B), 256, 0, stream>>>(
        Wvh, xh, bv, vh, C, N, 0, (long long)N * C, (long long)C * N, 1);

    // 2) energy = q * k  (fp32, into attention output region), then row softmax
    k_energy<<<dim3(N / 128, N / 64, B), 256, 0, stream>>>(
        qh, kth, attn, N, D, (long long)N * D, (long long)N * D, (long long)N * N);
    k_softmax<<<B * N, 256, 0, stream>>>(attn, N);

    // 3) out = gamma * (v * attn^T) + x
    k_out<<<dim3(N / 128, C / 64, B), 256, 0, stream>>>(
        attn, vh, x, gamma, out, N, C,
        (long long)N * N, (long long)C * N, (long long)C * N, (long long)C * N);
}
